// DAM_Fusion_Block_11836929868331
// MI455X (gfx1250) — matrix operand order repaired and checked by reading, not compiled, not run on hardware
//
#include <hip/hip_runtime.h>
#include <hip/hip_bf16.h>
#include <math.h>
#include <stdint.h>

// ---------------------------------------------------------------------------
// MI455X (gfx1250) DAM fusion block.
//  - GEMMs (1x1 convs + visual cross-attention) on v_wmma_f32_16x16x32_bf16; audio attention is scalar FMA
//  - LDS tiles staged in WMMA fragment layout; quad-packed ds_store_b64,
//    fragment reads are ds_load_b128 pairs
//  - Double-buffered K pipeline (stage k+1 while computing k, 1 barrier/step)
//  - 4-way fused conv over x; elementwise passes fused into GEMM epilogues
//  - InstanceNorm rows DMA'd to LDS via the Tensor Data Mover (TENSORcnt)
// ---------------------------------------------------------------------------

typedef __bf16 bf16;
typedef __attribute__((ext_vector_type(16))) __bf16 v16bf;
typedef __attribute__((ext_vector_type(4)))  __bf16 v4bf;
typedef __attribute__((ext_vector_type(8)))  float  v8f;
typedef __attribute__((ext_vector_type(4)))  float  v4f;
typedef __attribute__((ext_vector_type(4)))  unsigned int u32x4;
typedef __attribute__((ext_vector_type(4)))  int i32x4;
typedef __attribute__((ext_vector_type(8)))  int i32x8;

#define D_   256
#define BT_  64
#define HW_  4096
#define L_   32

enum { ACT_NONE = 0, ACT_GELU = 1, ACT_RELU = 2, ACT_TANH = 3 };
enum { EPI_STORE = 0, EPI_RES = 1, EPI_GATED = 2 };

#if __has_builtin(__builtin_amdgcn_tensor_load_to_lds) && \
    __has_builtin(__builtin_amdgcn_s_wait_tensorcnt)
#define HAVE_TDM 1
#else
#define HAVE_TDM 0
#endif

// ===========================================================================
// Pointwise-conv GEMM with fused staging/epilogue options.
//   out[b,o,t] = EPI( act( sum_c W[o,c] * (in[b,c,t] (*in2)) + bias[o] ) )
// Tile BM=64 x BN x BK=32; 8 waves (4xM, 2xN); double-buffered K pipeline.
// ===========================================================================
template <typename TIN, typename TOUT, int BN, int ACT, bool MASKMUL, bool MUL2, int EPI>
__global__ __launch_bounds__(256) void conv1_wmma(
    const TIN* __restrict__ in, const bf16* __restrict__ in2,
    const float* __restrict__ W, const float* __restrict__ bias,
    TOUT* __restrict__ out, const float* __restrict__ mask,
    const float* __restrict__ resid, const bf16* __restrict__ aux2, int Ltot)
{
    constexpr int BM = 64, BK = 32;
    constexpr int NACC = BN / 32;
    constexpr int NT   = BN / 16;
    constexpr int XQ   = (8 * BN) / 256;           // activation quads per thread
    __shared__ __align__(16) bf16 sWf[2][4][32][24];
    __shared__ __align__(16) bf16 sXf[2][NT][32][24];

    const int tid  = threadIdx.x;
    const int wave = tid >> 5;
    const int lane = tid & 31;
    const int wm   = wave & 3;
    const int wn   = wave >> 2;
    const int b     = blockIdx.z;
    const int mBase = blockIdx.y * BM;
    const int nBase = blockIdx.x * BN;

    v8f acc[NACC];
#pragma unroll
    for (int j = 0; j < NACC; ++j) { v8f z = {}; acc[j] = z; }

    const size_t inB = (size_t)b * D_ * (size_t)Ltot;

    auto stage = [&](int bufi, int kk) {
        // weights: 512 quads, 2 per thread, K-contiguous float4 -> 4x bf16
#pragma unroll
        for (int it = 0; it < 2; ++it) {
            const int i    = tid + (it << 8);
            const int wmi  = i >> 7;
            const int rem2 = i & 127;
            const int lam  = rem2 >> 2;
            const int h    = (rem2 & 3) << 2;
            const int row  = wmi * 16 + (lam & 15);
            const int kb   = h + ((lam >> 4) << 4);
            const v4f w = *(const v4f*)&W[(mBase + row) * D_ + kk + kb];
            v4bf p; p[0] = (bf16)w[0]; p[1] = (bf16)w[1]; p[2] = (bf16)w[2]; p[3] = (bf16)w[3];
            *(v4bf*)&sWf[bufi][wmi][lam][h] = p;
        }
        // activations: 4 row-strided coalesced loads -> one ds_store_b64
#pragma unroll
        for (int it = 0; it < XQ; ++it) {
            const int i   = tid + (it << 8);
            const int r   = (i / BN) << 2;
            const int c   = i & (BN - 1);
            const int j   = c >> 4;
            const int lam = (c & 15) + ((r >> 4) << 4);
            const int h   = r & 15;
            const size_t g = inB + (size_t)(kk + r) * Ltot + nBase + c;
            float v0 = (float)in[g];
            float v1 = (float)in[g + Ltot];
            float v2 = (float)in[g + 2 * (size_t)Ltot];
            float v3 = (float)in[g + 3 * (size_t)Ltot];
            if (MUL2) {
                v0 *= (float)in2[g];
                v1 *= (float)in2[g + Ltot];
                v2 *= (float)in2[g + 2 * (size_t)Ltot];
                v3 *= (float)in2[g + 3 * (size_t)Ltot];
            }
            v4bf p; p[0] = (bf16)v0; p[1] = (bf16)v1; p[2] = (bf16)v2; p[3] = (bf16)v3;
            *(v4bf*)&sXf[bufi][j][lam][h] = p;
        }
    };

    stage(0, 0);
    __syncthreads();
    for (int ks = 0; ks < 8; ++ks) {
        const int cur = ks & 1;
        if (ks < 7) stage(cur ^ 1, (ks + 1) << 5);
        const v16bf a = *(const v16bf*)&sWf[cur][wm][lane][0];
#pragma unroll
        for (int j = 0; j < NACC; ++j) {
            const v16bf bv = *(const v16bf*)&sXf[cur][wn * (NT / 2) + j][lane][0];
            acc[j] = __builtin_amdgcn_wmma_f32_16x16x32_bf16(
                false, a, false, bv, (short)0, acc[j], false, false);
        }
        __syncthreads();
    }

#pragma unroll
    for (int j = 0; j < NACC; ++j) {
        const int N = wn * (BN / 2) + j * 16 + (lane & 15);
        const int t = nBase + N;
        const float mk = MASKMUL ? mask[b * Ltot + t] : 1.0f;
#pragma unroll
        for (int r = 0; r < 8; ++r) {
            const int M = r + ((lane >> 4) << 3);
            const int o = mBase + wm * 16 + M;
            const size_t idx = (size_t)b * D_ * (size_t)Ltot + (size_t)o * Ltot + t;
            float v = acc[j][r];
            if (bias) v += bias[o];
            if (ACT == ACT_GELU)      v = 0.5f * v * (1.0f + erff(v * 0.70710678118f));
            else if (ACT == ACT_RELU) v = fmaxf(v, 0.0f);
            else if (ACT == ACT_TANH) v = tanhf(v);
            if (EPI == EPI_STORE)      out[idx] = (TOUT)(v * mk);
            else if (EPI == EPI_RES)   out[idx] = (TOUT)(resid[idx] + v);
            else /* EPI_GATED */       out[idx] = (TOUT)(resid[idx] + v * (float)aux2[idx]);
        }
    }
}

// ===========================================================================
// 4-way fused conv over x: stages each x tile ONCE, 4 GEMMs per tile.
// Set 0 applies GELU (vis_project); sets 1..3 raw (vs_q1, as_key, vv).
// ===========================================================================
__global__ __launch_bounds__(256) void conv1_wmma_x4(
    const float* __restrict__ in,
    const float* __restrict__ W0, const float* __restrict__ c0, bf16* __restrict__ o0,
    const float* __restrict__ W1, const float* __restrict__ c1, bf16* __restrict__ o1,
    const float* __restrict__ W2, const float* __restrict__ c2, bf16* __restrict__ o2,
    const float* __restrict__ W3, const float* __restrict__ c3, bf16* __restrict__ o3)
{
    constexpr int BM = 64, BN = 64, NACC = 2;
    __shared__ __align__(16) bf16 sWf[2][4][4][32][24];   // [buf][set][mtile][lane][h]
    __shared__ __align__(16) bf16 sXf[2][4][32][24];

    const int tid  = threadIdx.x;
    const int wave = tid >> 5;
    const int lane = tid & 31;
    const int wm   = wave & 3;
    const int wn   = wave >> 2;
    const int b     = blockIdx.z;
    const int mBase = blockIdx.y * BM;
    const int nBase = blockIdx.x * BN;

    v8f acc[4][NACC];
#pragma unroll
    for (int s = 0; s < 4; ++s)
#pragma unroll
        for (int j = 0; j < NACC; ++j) { v8f z = {}; acc[s][j] = z; }

    const size_t inB = (size_t)b * D_ * (size_t)HW_;

    auto stage = [&](int bufi, int kk) {
        // 4 weight sets x 512 quads = 8 unrolled iterations (set static per it)
#pragma unroll
        for (int it = 0; it < 8; ++it) {
            const float* Wp = (it < 2) ? W0 : (it < 4) ? W1 : (it < 6) ? W2 : W3;
            const int set  = it >> 1;
            const int i    = tid + ((it & 1) << 8);
            const int wmi  = i >> 7;
            const int rem2 = i & 127;
            const int lam  = rem2 >> 2;
            const int h    = (rem2 & 3) << 2;
            const int row  = wmi * 16 + (lam & 15);
            const int kb   = h + ((lam >> 4) << 4);
            const v4f w = *(const v4f*)&Wp[(mBase + row) * D_ + kk + kb];
            v4bf p; p[0] = (bf16)w[0]; p[1] = (bf16)w[1]; p[2] = (bf16)w[2]; p[3] = (bf16)w[3];
            *(v4bf*)&sWf[bufi][set][wmi][lam][h] = p;
        }
        // x tile: 512 quads, 2 per thread
#pragma unroll
        for (int it = 0; it < 2; ++it) {
            const int i   = tid + (it << 8);
            const int r   = (i / BN) << 2;
            const int c   = i & (BN - 1);
            const int j   = c >> 4;
            const int lam = (c & 15) + ((r >> 4) << 4);
            const int h   = r & 15;
            const size_t g = inB + (size_t)(kk + r) * HW_ + nBase + c;
            v4bf p;
            p[0] = (bf16)in[g];
            p[1] = (bf16)in[g + HW_];
            p[2] = (bf16)in[g + 2 * (size_t)HW_];
            p[3] = (bf16)in[g + 3 * (size_t)HW_];
            *(v4bf*)&sXf[bufi][j][lam][h] = p;
        }
    };

    stage(0, 0);
    __syncthreads();
    for (int ks = 0; ks < 8; ++ks) {
        const int cur = ks & 1;
        if (ks < 7) stage(cur ^ 1, (ks + 1) << 5);
        const v16bf a0 = *(const v16bf*)&sWf[cur][0][wm][lane][0];
        const v16bf a1 = *(const v16bf*)&sWf[cur][1][wm][lane][0];
        const v16bf a2 = *(const v16bf*)&sWf[cur][2][wm][lane][0];
        const v16bf a3 = *(const v16bf*)&sWf[cur][3][wm][lane][0];
#pragma unroll
        for (int j = 0; j < NACC; ++j) {
            const v16bf bv = *(const v16bf*)&sXf[cur][wn * 2 + j][lane][0];
            acc[0][j] = __builtin_amdgcn_wmma_f32_16x16x32_bf16(false, a0, false, bv, (short)0, acc[0][j], false, false);
            acc[1][j] = __builtin_amdgcn_wmma_f32_16x16x32_bf16(false, a1, false, bv, (short)0, acc[1][j], false, false);
            acc[2][j] = __builtin_amdgcn_wmma_f32_16x16x32_bf16(false, a2, false, bv, (short)0, acc[2][j], false, false);
            acc[3][j] = __builtin_amdgcn_wmma_f32_16x16x32_bf16(false, a3, false, bv, (short)0, acc[3][j], false, false);
        }
        __syncthreads();
    }

    const float* biases[4] = {c0, c1, c2, c3};
    bf16*        outs[4]   = {o0, o1, o2, o3};
#pragma unroll
    for (int s = 0; s < 4; ++s) {
#pragma unroll
        for (int j = 0; j < NACC; ++j) {
            const int N = wn * 32 + j * 16 + (lane & 15);
            const int t = nBase + N;
#pragma unroll
            for (int r = 0; r < 8; ++r) {
                const int M = r + ((lane >> 4) << 3);
                const int o = mBase + wm * 16 + M;
                float v = acc[s][j][r] + biases[s][o];
                if (s == 0) v = 0.5f * v * (1.0f + erff(v * 0.70710678118f));  // GELU
                outs[s][(size_t)b * D_ * (size_t)HW_ + (size_t)o * HW_ + t] = (bf16)v;
            }
        }
    }
}

// ===========================================================================
// InstanceNorm1d in place; one block per (b,c) row; TDM DMA when available.
// ===========================================================================
template <int LEN>
__global__ __launch_bounds__(256) void inorm_ip(bf16* __restrict__ buf)
{
    __shared__ __align__(16) bf16 row[LEN];
    __shared__ float ss[256], sq[256];
    const size_t base = (size_t)blockIdx.x * LEN;

#if HAVE_TDM
    if (threadIdx.x < 32) {
        const uint64_t ga = (uint64_t)(uintptr_t)(buf + base);
        const uint32_t ldsoff = (uint32_t)(uintptr_t)(&row[0]);
        u32x4 g0;
        g0[0] = 1u;
        g0[1] = ldsoff;
        g0[2] = (uint32_t)ga;
        g0[3] = (uint32_t)((ga >> 32) & 0x01FFFFFFu) | (2u << 30);
        i32x8 g1;
        g1[0] = 0x00010000;                                            // data_size=2B
        g1[1] = (int)(((uint32_t)LEN & 0xFFFFu) << 16);                // tensor_dim0 lo
        g1[2] = (int)((((uint32_t)LEN >> 16) & 0xFFFFu) | (1u << 16)); // dim0 hi | dim1=1
        g1[3] = (int)(((uint32_t)LEN & 0xFFFFu) << 16);                // tile_dim0
        g1[4] = 0;
        g1[5] = (int)LEN;                                              // dim0 stride
        g1[6] = 0; g1[7] = 0;
        i32x4 z4 = {0, 0, 0, 0};
#if __has_include(<hip/amd_detail/amd_gfx1250_TDM.h>)
        i32x8 z8 = {0, 0, 0, 0, 0, 0, 0, 0};
        __builtin_amdgcn_tensor_load_to_lds(g0, g1, z4, z4, z8, 0);
#else
        __builtin_amdgcn_tensor_load_to_lds(g0, g1, z4, z4, 0);
#endif
        __builtin_amdgcn_s_wait_tensorcnt(0);
    }
    __syncthreads();
#else
    for (int i = threadIdx.x; i < LEN; i += 256) row[i] = buf[base + i];
    __syncthreads();
#endif

    float s = 0.f, q = 0.f;
    for (int i = threadIdx.x; i < LEN; i += 256) {
        const float v = (float)row[i];
        s += v; q += v * v;
    }
    ss[threadIdx.x] = s; sq[threadIdx.x] = q;
    __syncthreads();
    for (int off = 128; off > 0; off >>= 1) {
        if ((int)threadIdx.x < off) {
            ss[threadIdx.x] += ss[threadIdx.x + off];
            sq[threadIdx.x] += sq[threadIdx.x + off];
        }
        __syncthreads();
    }
    const float mean = ss[0] / (float)LEN;
    float var = sq[0] / (float)LEN - mean * mean;
    var = fmaxf(var, 0.f);
    const float inv = rsqrtf(var + 1e-5f);
    for (int i = threadIdx.x; i < LEN; i += 256)
        buf[base + i] = (bf16)(((float)row[i] - mean) * inv);
}

// ===========================================================================
// Visual cross-attention (fused logits->softmax->PV, all WMMA)
// ===========================================================================
__global__ __launch_bounds__(256) void attn_v_wmma(
    const bf16* __restrict__ q, const bf16* __restrict__ kk,
    const bf16* __restrict__ av, const float* __restrict__ amask,
    const float* __restrict__ visw, bf16* __restrict__ outtok)
{
    const int tid = threadIdx.x, wave = tid >> 5, lane = tid & 31;
    const int b = blockIdx.y;
    const int tTile = blockIdx.x * 128 + wave * 16;
    __shared__ __align__(16) bf16 sp[8][32][24];

    v8f acc0 = {}, acc1 = {};
    const size_t qB = (size_t)b * D_ * HW_;
    const size_t kB = (size_t)b * D_ * L_;

    const int rowT = tTile + (lane & 15);
    const int kh   = (lane >> 4) << 3;
    const int kb16 = (lane >> 4) << 4;
    const int nlo  = lane & 15;

    for (int c0 = 0; c0 < D_; c0 += 32) {
        v16bf a;
#pragma unroll
        for (int h = 0; h < 16; ++h) {
            const int c = c0 + kb16 + h;
            a[h] = q[qB + (size_t)c * HW_ + rowT];
        }
        v16bf b0, b1;
#pragma unroll
        for (int h = 0; h < 16; ++h) {
            const int c = c0 + kb16 + h;
            b0[h] = kk[kB + (size_t)c * L_ + nlo];
            b1[h] = kk[kB + (size_t)c * L_ + 16 + nlo];
        }
        acc0 = __builtin_amdgcn_wmma_f32_16x16x32_bf16(false, a, false, b0, (short)0, acc0, false, false);
        acc1 = __builtin_amdgcn_wmma_f32_16x16x32_bf16(false, a, false, b1, (short)0, acc1, false, false);
    }

    const float scale = 0.0625f;
    const float mt0 = 10000.f * amask[b * L_ + nlo] - 10000.f;
    const float mt1 = 10000.f * amask[b * L_ + 16 + nlo] - 10000.f;
    const float vw = expf(visw[0]) / expf(visw[0]);

#pragma unroll
    for (int r = 0; r < 8; ++r) {
        float x0 = acc0[r] * scale + mt0;
        float x1 = acc1[r] * scale + mt1;
        float m = fmaxf(x0, x1);
#pragma unroll
        for (int off = 1; off < 16; off <<= 1) m = fmaxf(m, __shfl_xor(m, off, 32));
        float e0 = expf(x0 - m), e1 = expf(x1 - m);
        float s = e0 + e1;
#pragma unroll
        for (int off = 1; off < 16; off <<= 1) s += __shfl_xor(s, off, 32);
        const float inv = vw / s;
        const int M = r + ((lane >> 4) << 3);
        const int l0 = nlo;
        const int h0 = (l0 & 7) + ((l0 & 16) >> 1);
        sp[wave][(M & 15) + (((l0 >> 3) & 1) << 4)][h0] = (bf16)(e0 * inv);
        const int l1 = nlo + 16;
        const int h1 = (l1 & 7) + ((l1 & 16) >> 1);
        sp[wave][(M & 15) + (((l1 >> 3) & 1) << 4)][h1] = (bf16)(e1 * inv);
    }
    __syncthreads();

    const size_t oB = (size_t)b * D_ * HW_;
    const v16bf ap = *(const v16bf*)&sp[wave][lane][0];
    for (int ct = 0; ct < 16; ++ct) {
        v16bf bv;
#pragma unroll
        for (int h = 0; h < 16; ++h) {
            const int l = ((h & 8) << 1) + (h & 7) + kh;
            bv[h] = av[kB + (size_t)(ct * 16 + nlo) * L_ + l];
        }
        v8f d = {};
        d = __builtin_amdgcn_wmma_f32_16x16x32_bf16(false, ap, false, bv, (short)0, d, false, false);
#pragma unroll
        for (int r = 0; r < 8; ++r) {
            const int M = r + ((lane >> 4) << 3);
            outtok[oB + (size_t)(ct * 16 + nlo) * HW_ + tTile + M] = (bf16)d[r];
        }
    }
}

// ===========================================================================
// Audio-query attention (softmax over HW in LDS)
// ===========================================================================
__global__ __launch_bounds__(256) void attn_a_kernel(
    const bf16* __restrict__ aq, const bf16* __restrict__ ak,
    const bf16* __restrict__ vv, const float* __restrict__ amask,
    const float* __restrict__ audw, bf16* __restrict__ audioo)
{
    const int b = blockIdx.x >> 5;
    const int l = blockIdx.x & 31;
    const int tid = threadIdx.x;
    __shared__ float srow[256];
    __shared__ float sp[HW_];
    __shared__ float red[256];

    const float mrow = amask[b * L_ + l];
    srow[tid] = (float)aq[((size_t)b * D_ + tid) * L_ + l] * mrow;
    __syncthreads();

    const float scale = 0.0625f;
    const float mterm = 10000.f * mrow - 10000.f;
    const size_t akB = (size_t)b * D_ * HW_;

    float lmax = -3.0e38f;
    for (int t = tid; t < HW_; t += 256) {
        float dot = 0.f;
        for (int c = 0; c < D_; ++c)
            dot += srow[c] * (float)ak[akB + (size_t)c * HW_ + t];
        dot = dot * scale + mterm;
        sp[t] = dot;
        lmax = fmaxf(lmax, dot);
    }
    red[tid] = lmax; __syncthreads();
    for (int off = 128; off > 0; off >>= 1) {
        if (tid < off) red[tid] = fmaxf(red[tid], red[tid + off]);
        __syncthreads();
    }
    const float gmax = red[0];
    __syncthreads();
    float lsum = 0.f;
    for (int t = tid; t < HW_; t += 256) {
        const float e = expf(sp[t] - gmax);
        sp[t] = e; lsum += e;
    }
    red[tid] = lsum; __syncthreads();
    for (int off = 128; off > 0; off >>= 1) {
        if (tid < off) red[tid] += red[tid + off];
        __syncthreads();
    }
    const float aw  = expf(audw[0]) / expf(audw[0]);
    const float inv = aw / red[0];
    __syncthreads();

    float acc = 0.f;
    const size_t vvB = (size_t)b * D_ * HW_ + (size_t)tid * HW_;
    for (int t = 0; t < HW_; ++t)
        acc += sp[t] * (float)vv[vvB + t];
    audioo[((size_t)b * D_ + tid) * L_ + l] = (bf16)(acc * inv);
}

// ===========================================================================
extern "C" void kernel_launch(void* const* d_in, const int* in_sizes, int n_in,
                              void* d_out, int out_size, void* d_ws, size_t ws_size,
                              hipStream_t stream)
{
    (void)in_sizes; (void)n_in; (void)out_size; (void)ws_size;

    const float* x     = (const float*)d_in[0];
    const float* audio = (const float*)d_in[1];
    const float* amask = (const float*)d_in[2];
    const float* vp_w  = (const float*)d_in[3];  const float* vp_b  = (const float*)d_in[4];
    const float* av_w  = (const float*)d_in[5];  const float* av_b  = (const float*)d_in[6];
    const float* vv_w  = (const float*)d_in[7];  const float* vv_b  = (const float*)d_in[8];
    const float* vo_w  = (const float*)d_in[9];  const float* vo_b  = (const float*)d_in[10];
    const float* ao_w  = (const float*)d_in[11]; const float* ao_b  = (const float*)d_in[12];
    const float* vsk_w = (const float*)d_in[13]; const float* vsk_b = (const float*)d_in[14];
    const float* vq1_w = (const float*)d_in[15]; const float* vq1_b = (const float*)d_in[16];
    const float* vq2_w = (const float*)d_in[17]; const float* vq2_b = (const float*)d_in[18];
    const float* ask_w = (const float*)d_in[19]; const float* ask_b = (const float*)d_in[20];
    const float* aq1_w = (const float*)d_in[21]; const float* aq1_b = (const float*)d_in[22];
    const float* aq2_w = (const float*)d_in[23]; const float* aq2_b = (const float*)d_in[24];
    const float* oc_w  = (const float*)d_in[25]; const float* oc_b  = (const float*)d_in[26];
    const float* aoc_w = (const float*)d_in[27]; const float* aoc_b = (const float*)d_in[28];
    const float* g1_w  = (const float*)d_in[29];
    const float* g2_w  = (const float*)d_in[30];
    const float* visw  = (const float*)d_in[31];
    const float* audw  = (const float*)d_in[32];

    const size_t NBIG = (size_t)BT_ * D_ * HW_;
    const size_t NSML = (size_t)BT_ * D_ * L_;

    bf16* B0 = (bf16*)d_ws;          // vis
    bf16* B1 = B0 + NBIG;            // q1 -> outtok -> vis2
    bf16* B2 = B1 + NBIG;            // ak -> gate_pre
    bf16* B3 = B2 + NBIG;            // vv
    bf16* B4 = B3 + NBIG;            // q -> vis_out
    bf16* avb = B4 + NBIG;
    bf16* kb  = avb + NSML;
    bf16* aqt = kb + NSML;
    bf16* aqb = aqt + NSML;
    bf16* aob = aqb + NSML;
    bf16* arb = aob + NSML;

    float* xout = (float*)d_out;
    float* aout = xout + NBIG;

    dim3 blk(256);
    dim3 gVis(HW_ / 128, D_ / 64, BT_);
    dim3 gX4 (HW_ / 64,  D_ / 64, BT_);
    dim3 gAud(1, D_ / 64, BT_);
    const int gNorm = BT_ * D_;
    const bf16*  nb = nullptr;
    const float* nf = nullptr;

    // ---- 4 fused GEMMs from x: vis(GELU), q1, as_key, vv ----
    conv1_wmma_x4<<<gX4, blk, 0, stream>>>(x,
        vp_w, vp_b, B0,  vq1_w, vq1_b, B1,  ask_w, ask_b, B2,  vv_w, vv_b, B3);
    inorm_ip<HW_><<<gNorm, blk, 0, stream>>>(B1);
    inorm_ip<HW_><<<gNorm, blk, 0, stream>>>(B2);                            // ak
    inorm_ip<HW_><<<gNorm, blk, 0, stream>>>(B3);                            // vv
    conv1_wmma<bf16, bf16, 128, ACT_NONE, false, false, EPI_STORE><<<gVis, blk, 0, stream>>>(
        B1, nb, vq2_w, vq2_b, B4, nf, nf, nb, HW_);
    inorm_ip<HW_><<<gNorm, blk, 0, stream>>>(B4);                            // q

    // ---- audio-side GEMMs ----
    conv1_wmma<float, bf16, 32, ACT_NONE, true , false, EPI_STORE><<<gAud, blk, 0, stream>>>(
        audio, nb, av_w, av_b, avb, amask, nf, nb, L_);
    conv1_wmma<float, bf16, 32, ACT_NONE, true , false, EPI_STORE><<<gAud, blk, 0, stream>>>(
        audio, nb, vsk_w, vsk_b, kb, amask, nf, nb, L_);
    conv1_wmma<float, bf16, 32, ACT_NONE, false, false, EPI_STORE><<<gAud, blk, 0, stream>>>(
        audio, nb, aq1_w, aq1_b, aqt, nf, nf, nb, L_);
    conv1_wmma<bf16, bf16, 32, ACT_NONE, false, false, EPI_STORE><<<gAud, blk, 0, stream>>>(
        aqt, nb, aq2_w, aq2_b, aqb, nf, nf, nb, L_);

    // ---- cross-attentions ----
    attn_v_wmma<<<dim3(HW_ / 128, BT_), blk, 0, stream>>>(B4, kb, avb, amask, visw, B1);
    conv1_wmma<bf16, bf16, 128, ACT_NONE, false, false, EPI_STORE><<<gVis, blk, 0, stream>>>(
        B1, nb, vo_w, vo_b, B4, nf, nf, nb, HW_);
    inorm_ip<HW_><<<gNorm, blk, 0, stream>>>(B4);                            // vis_out

    attn_a_kernel<<<BT_ * L_, blk, 0, stream>>>(aqb, B2, B3, amask, audw, aob);
    conv1_wmma<bf16, bf16, 32, ACT_NONE, false, false, EPI_STORE><<<gAud, blk, 0, stream>>>(
        aob, nb, ao_w, ao_b, arb, nf, nf, nb, L_);

    // ---- fusion + gated residual (elementwise fused into epilogues) ----
    conv1_wmma<bf16, bf16, 128, ACT_GELU, false, true, EPI_STORE><<<gVis, blk, 0, stream>>>(
        B4, B0, oc_w, oc_b, B1, nf, nf, nb, HW_);                            // vis2
    conv1_wmma<bf16, bf16, 128, ACT_RELU, false, false, EPI_STORE><<<gVis, blk, 0, stream>>>(
        B1, nb, g1_w, nullptr, B2, nf, nf, nb, HW_);                         // gate_pre
    conv1_wmma<bf16, float, 128, ACT_TANH, false, false, EPI_GATED><<<gVis, blk, 0, stream>>>(
        B2, nb, g2_w, nullptr, xout, nf, x, B1, HW_);                        // xout = x + tanh(..)*vis2

    // ---- audio output ----
    conv1_wmma<bf16, float, 32, ACT_GELU, false, false, EPI_RES><<<gAud, blk, 0, stream>>>(
        arb, nb, aoc_w, aoc_b, aout, nf, audio, nb, L_);                     // aout = audio + gelu(..)
}
